// Decoder_1322849927224
// MI455X (gfx1250) — matrix operand order repaired and checked by reading, not compiled, not run on hardware
//
#include <hip/hip_runtime.h>

// ---------------- types ----------------
typedef __bf16 bf16;
typedef __attribute__((ext_vector_type(16))) __bf16 bf16x16;
typedef __attribute__((ext_vector_type(8)))  __bf16 bf16x8;
typedef __attribute__((ext_vector_type(8)))  float  f32x8;

#define B_   512
#define H_   1024
#define G4H_ 4096
#define T_   128

__device__ __forceinline__ float sigmoidf_(float x) {
    return 1.0f / (1.0f + __expf(-x));
}

// CDNA5 async global->LDS copy (16B per lane), tracked by ASYNCcnt.
__device__ __forceinline__ void async_copy_b128(unsigned lds_off, const void* gptr) {
    asm volatile("global_load_async_to_lds_b128 %0, %1, off"
                 :: "v"(lds_off), "v"((unsigned long long)(uintptr_t)gptr)
                 : "memory");
}
__device__ __forceinline__ void wait_async0() {
    asm volatile("s_wait_asynccnt 0x0" ::: "memory");
}
__device__ __forceinline__ unsigned lds_addr(const void* p) {
    // generic LDS pointer: low 32 bits are the LDS byte offset (aperture scheme)
    return (unsigned)(uintptr_t)p;
}

// ---------------- setup kernels ----------------
__global__ void cvt_f32_to_bf16(const float* __restrict__ src, bf16* __restrict__ dst, int n) {
    int i = blockIdx.x * blockDim.x + threadIdx.x;
    if (i < n) dst[i] = (bf16)src[i];
}

__global__ void bias_sum_kernel(const float* __restrict__ a, const float* __restrict__ b,
                                float* __restrict__ o, int n) {
    int i = blockIdx.x * blockDim.x + threadIdx.x;
    if (i < n) o[i] = a[i] + b[i];
}

__global__ void zero_f32(float* __restrict__ p, int n) {
    int i = blockIdx.x * blockDim.x + threadIdx.x;
    if (i < n) p[i] = 0.0f;
}

// ---------------- fused LSTM gates + cell kernel ----------------
// gates[b, g*H+j] = sum_seg A_seg[b,:] . W_seg[g*H+j,:]  (bf16 WMMA, fp32 acc)
//                 + bsum[g*H+j] + (xin ? xin[b]*wih_col[g*H+j] : 0)
// then the LSTM cell update -> c_out (fp32), h_out (bf16).
// Grid: (B/64, H/64); block 256 threads (8 waves).
// Wave (waveM in 0..1, waveN in 0..3): 32 rows x 16 hidden x 4 gates = 8 WMMA tiles.
// Double-buffered LDS fed by global_load_async_to_lds_b128.
__global__ __launch_bounds__(256, 2)
void lstm_gates_kernel(const bf16* __restrict__ A0, const bf16* __restrict__ W0,
                       const bf16* __restrict__ A1, const bf16* __restrict__ W1,
                       const float* __restrict__ bsum,
                       const float* __restrict__ xin, const float* __restrict__ wih_col,
                       const float* __restrict__ c_in,
                       float* __restrict__ c_out, bf16* __restrict__ h_out)
{
    __shared__ __align__(64) bf16 lsA[2][64 * 32];       // [buf][m][k]        2 x 4 KB
    __shared__ __align__(64) bf16 lsB[2][4 * 64 * 32];   // [buf][gate][n][k]  2 x 16 KB

    const int tid   = threadIdx.x;
    const int lane  = tid & 31;
    const int wv    = tid >> 5;
    const int waveN = wv & 3;          // 0..3 -> 16-col subtile
    const int waveM = wv >> 2;         // 0..1 -> 32-row subtile
    const int mBlock = blockIdx.x * 64;
    const int nBlock = blockIdx.y * 64;
    const int hf  = lane >> 4;         // half-wave
    const int l16 = lane & 15;

    f32x8 acc[2][4];
    #pragma unroll
    for (int s = 0; s < 2; ++s)
        #pragma unroll
        for (int g = 0; g < 4; ++g) acc[s][g] = (f32x8){};

    // staging split (256 threads): A 16B/thread, B 64B/thread
    const int ar = tid >> 2;               // A row 0..63
    const int ak = (tid & 3) * 8;          // A k-chunk {0,8,16,24}
    const int bg = tid >> 6;               // gate 0..3
    const int bn = tid & 63;               // hidden unit 0..63

    const int nIter = (A1 != nullptr) ? 64 : 32;  // 32 k-iters per 1024-K segment

    auto stage = [&](int it, int buf) {
        const bf16* __restrict__ A = (it >= 32) ? A1 : A0;
        const bf16* __restrict__ W = (it >= 32) ? W1 : W0;
        const int k0 = (it & 31) * 32;
        async_copy_b128(lds_addr(&lsA[buf][ar * 32 + ak]),
                        &A[(mBlock + ar) * H_ + k0 + ak]);
        const bf16* gw = &W[(bg * H_ + nBlock + bn) * H_ + k0];
        bf16* lw = &lsB[buf][(bg * 64 + bn) * 32];
        #pragma unroll
        for (int c = 0; c < 4; ++c)
            async_copy_b128(lds_addr(lw + c * 8), gw + c * 8);
    };

    // prologue: stage tile 0
    stage(0, 0);
    wait_async0();
    __syncthreads();

    for (int it = 0; it < nIter; ++it) {
        const int buf = it & 1;
        if (it + 1 < nIter) stage(it + 1, buf ^ 1);   // overlap next copy with compute

        // ---- A fragments (16x32 bf16 layout): lanes 0-15 K{0..15},
        //      lanes 16-31 K{16..31} for the same 16 rows ----
        bf16x16 afrag[2];
        #pragma unroll
        for (int s = 0; s < 2; ++s) {
            const int arow = (waveM * 32 + s * 16 + l16) * 32;
            bf16x8 alo = *(const bf16x8*)(&lsA[buf][arow + hf * 16]);
            bf16x8 ahi = *(const bf16x8*)(&lsA[buf][arow + hf * 16 + 8]);
            #pragma unroll
            for (int i = 0; i < 8; ++i) { afrag[s][i] = alo[i]; afrag[s][8 + i] = ahi[i]; }
        }
        // ---- B fragments (32x16): lane n = column, halves split K 0..15 / 16..31 ----
        bf16x16 bfrag[4];
        #pragma unroll
        for (int g = 0; g < 4; ++g)
            bfrag[g] = *(const bf16x16*)(&lsB[buf][(g * 64 + waveN * 16 + l16) * 32 + hf * 16]);

        #pragma unroll
        for (int s = 0; s < 2; ++s)
            #pragma unroll
            for (int g = 0; g < 4; ++g)
                acc[s][g] = __builtin_amdgcn_wmma_f32_16x16x32_bf16(
                    false, afrag[s], false, bfrag[g], (short)0, acc[s][g], false, false);

        if (it + 1 < nIter) {
            wait_async0();      // next tile fully landed in LDS (this wave's copies)
            __syncthreads();    // all waves' copies landed; safe to switch buffers
        }
    }

    // ---- epilogue: bias + input term + LSTM cell (gate order i,f,g,o) ----
    const int j = nBlock + waveN * 16 + l16;
    const float bi  = bsum[0 * H_ + j];
    const float bf  = bsum[1 * H_ + j];
    const float bg_ = bsum[2 * H_ + j];
    const float bo  = bsum[3 * H_ + j];
    float wi = 0.f, wf = 0.f, wg = 0.f, wo = 0.f;
    if (wih_col) {
        wi = wih_col[0 * H_ + j]; wf = wih_col[1 * H_ + j];
        wg = wih_col[2 * H_ + j]; wo = wih_col[3 * H_ + j];
    }
    #pragma unroll
    for (int s = 0; s < 2; ++s) {
        const int mBase = mBlock + waveM * 32 + s * 16 + hf * 8;  // C: VGPR r -> M = mBase+r
        #pragma unroll
        for (int r = 0; r < 8; ++r) {
            const int m = mBase + r;
            const float xv = xin ? xin[m] : 0.0f;
            const float gi = acc[s][0][r] + bi  + xv * wi;
            const float gf = acc[s][1][r] + bf  + xv * wf;
            const float gg = acc[s][2][r] + bg_ + xv * wg;
            const float go = acc[s][3][r] + bo  + xv * wo;
            const float iv = sigmoidf_(gi);
            const float fv = sigmoidf_(gf);
            const float gv = tanhf(gg);
            const float ov = sigmoidf_(go);
            const float c  = c_in[m * H_ + j];
            const float cn = fv * c + iv * gv;
            const float hn = ov * tanhf(cn);
            c_out[m * H_ + j] = cn;
            h_out[m * H_ + j] = (bf16)hn;
        }
    }
}

// ---------------- output projection + MSE loss ----------------
__global__ __launch_bounds__(256)
void out_loss_kernel(const bf16* __restrict__ h1, const float* __restrict__ Wout,
                     const float* __restrict__ bout, const float* __restrict__ outputs,
                     int t, float* __restrict__ xnext, float* __restrict__ loss)
{
    const int lane = threadIdx.x & 31;
    const int wv   = threadIdx.x >> 5;
    const int row  = blockIdx.x * 8 + wv;
    float s = 0.0f;
    for (int k = lane; k < H_; k += 32)
        s += (float)h1[row * H_ + k] * Wout[k];
    #pragma unroll
    for (int off = 16; off > 0; off >>= 1)
        s += __shfl_xor(s, off, 32);
    if (lane == 0) {
        const float o = s + bout[0];
        xnext[row] = o;
        const float d = o - outputs[row * T_ + t];
        atomicAdd(loss, d * d * (1.0f / (float)B_));
    }
}

// ---------------- host launcher ----------------
extern "C" void kernel_launch(void* const* d_in, const int* in_sizes, int n_in,
                              void* d_out, int out_size, void* d_ws, size_t ws_size,
                              hipStream_t stream) {
    (void)in_sizes; (void)n_in; (void)out_size; (void)ws_size;

    const float* outputs = (const float*)d_in[0];   // [512,128]
    const float* h0      = (const float*)d_in[1];   // [2,512,1024]
    const float* c0      = (const float*)d_in[2];   // [2,512,1024]
    const float* W_ih0   = (const float*)d_in[3];   // [4096,1]
    const float* W_hh0   = (const float*)d_in[4];   // [4096,1024]
    const float* b_ih0   = (const float*)d_in[5];
    const float* b_hh0   = (const float*)d_in[6];
    const float* W_ih1   = (const float*)d_in[7];   // [4096,1024]
    const float* W_hh1   = (const float*)d_in[8];   // [4096,1024]
    const float* b_ih1   = (const float*)d_in[9];
    const float* b_hh1   = (const float*)d_in[10];
    const float* W_out   = (const float*)d_in[11];  // [1,1024]
    const float* b_out   = (const float*)d_in[12];  // [1]
    float* lossp = (float*)d_out;

    // ---- workspace carve-out (256B aligned) ----
    char* ws = (char*)d_ws;
    size_t off = 0;
    auto carve = [&](size_t bytes) -> char* {
        char* p = ws + off;
        off += (bytes + 255) & ~(size_t)255;
        return p;
    };
    const size_t wsz = (size_t)G4H_ * H_ * sizeof(bf16);   // 8 MB
    const size_t hsz = (size_t)B_ * H_ * sizeof(bf16);     // 1 MB
    const size_t csz = (size_t)B_ * H_ * sizeof(float);    // 2 MB
    bf16* Whh0b = (bf16*)carve(wsz);
    bf16* Wih1b = (bf16*)carve(wsz);
    bf16* Whh1b = (bf16*)carve(wsz);
    bf16* hL0[2] = {(bf16*)carve(hsz), (bf16*)carve(hsz)};
    bf16* hL1[2] = {(bf16*)carve(hsz), (bf16*)carve(hsz)};
    float* cL0[2] = {(float*)carve(csz), (float*)carve(csz)};
    float* cL1[2] = {(float*)carve(csz), (float*)carve(csz)};
    float* bs0  = (float*)carve(G4H_ * sizeof(float));
    float* bs1  = (float*)carve(G4H_ * sizeof(float));
    float* xbuf = (float*)carve(B_ * sizeof(float));

    // ---- setup: weight/state conversion, bias sums, zero init ----
    {
        const int nW = G4H_ * H_;
        cvt_f32_to_bf16<<<(nW + 255) / 256, 256, 0, stream>>>(W_hh0, Whh0b, nW);
        cvt_f32_to_bf16<<<(nW + 255) / 256, 256, 0, stream>>>(W_ih1, Wih1b, nW);
        cvt_f32_to_bf16<<<(nW + 255) / 256, 256, 0, stream>>>(W_hh1, Whh1b, nW);
        const int nS = B_ * H_;
        cvt_f32_to_bf16<<<(nS + 255) / 256, 256, 0, stream>>>(h0,      hL0[0], nS);
        cvt_f32_to_bf16<<<(nS + 255) / 256, 256, 0, stream>>>(h0 + nS, hL1[0], nS);
        hipMemcpyAsync(cL0[0], c0,      csz, hipMemcpyDeviceToDevice, stream);
        hipMemcpyAsync(cL1[0], c0 + nS, csz, hipMemcpyDeviceToDevice, stream);
        bias_sum_kernel<<<(G4H_ + 255) / 256, 256, 0, stream>>>(b_ih0, b_hh0, bs0, G4H_);
        bias_sum_kernel<<<(G4H_ + 255) / 256, 256, 0, stream>>>(b_ih1, b_hh1, bs1, G4H_);
        zero_f32<<<(B_ + 255) / 256, 256, 0, stream>>>(xbuf, B_);
        zero_f32<<<1, 64, 0, stream>>>(lossp, 1);
    }

    // ---- time scan: layer0 -> layer1 -> output/loss per step ----
    const dim3 ggrid(B_ / 64, H_ / 64);  // (8, 16)
    for (int t = 0; t < T_; ++t) {
        const int p = t & 1, q = p ^ 1;
        // layer 0: gates = h0_prev @ W_hh0^T + x*W_ih0^T + biases
        lstm_gates_kernel<<<ggrid, 256, 0, stream>>>(
            hL0[p], Whh0b, nullptr, nullptr, bs0, xbuf, W_ih0,
            cL0[p], cL0[q], hL0[q]);
        // layer 1: gates = h0_new @ W_ih1^T + h1_prev @ W_hh1^T + biases (fused dual-K)
        lstm_gates_kernel<<<ggrid, 256, 0, stream>>>(
            hL0[q], Wih1b, hL1[p], Whh1b, bs1, nullptr, nullptr,
            cL1[p], cL1[q], hL1[q]);
        // output projection + feedback x + MSE loss accumulation
        out_loss_kernel<<<B_ / 8, 256, 0, stream>>>(
            hL1[q], W_out, b_out, outputs, t, xbuf, lossp);
    }
}
